// ModulatedConv2d_18786186953337
// MI455X (gfx1250) — matrix operand order repaired and checked by reading, not compiled, not run on hardware
//
#include <hip/hip_runtime.h>
#include <hip/hip_bf16.h>
#include <stddef.h>

typedef _Float16 v8h  __attribute__((ext_vector_type(8)));
typedef _Float16 v16h __attribute__((ext_vector_type(16)));
typedef float    v8f  __attribute__((ext_vector_type(8)));

// Problem constants (from reference setup_inputs)
#define B_    8
#define CIN   512
#define COUT  512
#define HW    64
#define NPOS  (HW*HW)        // 4096
#define KS    9              // 3x3 taps
#define KTOT  (CIN*KS)       // 4608
#define C_EQ  0.0147314057f  // 1/sqrt(512*9)
#define EPSV  1e-8f

// Workspace layout (halves):
//   xh[b][i][h][w] : B_*CIN*NPOS    = 16,777,216 halves (33.5 MB)
//   wm[b][r][o][i] : B_*KS*COUT*CIN = 18,874,368 halves (37.7 MB)
#define XH_HALFS ((size_t)B_ * CIN * NPOS)

// ---------------------------------------------------------------------------
// Kernel 1: modulate + demodulate -> f16 weights, tap-major layout wm[b][r][o][i]
// one block (256 thr) per (b,o) pair
// ---------------------------------------------------------------------------
__global__ void modulate_kernel(const float* __restrict__ w,
                                const float* __restrict__ s,
                                _Float16* __restrict__ wm) {
    const int bo = blockIdx.x;
    const int o  = bo & (COUT - 1);
    const int bi = bo >> 9;          // COUT==512
    const int tid = threadIdx.x;

    const float* wrow = w + (size_t)o * KTOT;   // [i][kh][kw] contiguous
    const float* srow = s + (size_t)bi * CIN;

    float sum = 0.0f;
    for (int e = tid; e < KTOT; e += 256) {
        const int i = e / 9;
        const float v = wrow[e] * srow[i];
        sum += v * v;
    }
    __shared__ float red[256];
    red[tid] = sum;
    __syncthreads();
    #pragma unroll
    for (int st = 128; st > 0; st >>= 1) {
        if (tid < st) red[tid] += red[tid + st];
        __syncthreads();
    }
    const float total = red[0];
    const float scale = C_EQ * rsqrtf(C_EQ * C_EQ * total + EPSV);

    for (int e = tid; e < KTOT; e += 256) {
        const int i = e / 9;
        const int r = e - i * 9;
        const float v = wrow[e] * srow[i] * scale;
        wm[(((size_t)bi * KS + r) * COUT + o) * CIN + i] = (_Float16)v;
    }
}

// ---------------------------------------------------------------------------
// Kernel 2: x (f32) -> xh (f16), elementwise, 4 per thread
// ---------------------------------------------------------------------------
__global__ void cvt_x_kernel(const float* __restrict__ x,
                             _Float16* __restrict__ xh) {
    const size_t i = ((size_t)blockIdx.x * 256 + threadIdx.x) * 4;
    const float4 v = *(const float4*)(x + i);
    _Float16* d = xh + i;
    d[0] = (_Float16)v.x; d[1] = (_Float16)v.y;
    d[2] = (_Float16)v.z; d[3] = (_Float16)v.w;
}

// ---------------------------------------------------------------------------
// Kernel 3: implicit-GEMM modulated conv via v_wmma_f32_16x16x32_f16
//   grid = (NPOS/128, COUT/256, B_), block = 256 (8 waves)
//   block tile: M=256 (Cout) x N=128 (hw), K-step 32, K loop = 9 taps x 16
//   wave tile: 64x64 -> 4x4 WMMA accumulators (16 wmma / K-step / wave)
// ---------------------------------------------------------------------------
__global__ __launch_bounds__(256)
void modconv_wmma_kernel(const _Float16* __restrict__ xh,
                         const _Float16* __restrict__ wm,
                         const float* __restrict__ bias,
                         float* __restrict__ out) {
    __shared__ _Float16 As[256][32];   // [m][k] k-contiguous, row = 64B (16 KB)
    __shared__ _Float16 Bs[128][32];   // [n][k] (B transposed), row = 64B (8 KB)

    const int tid  = threadIdx.x;
    const int lane = tid & 31;
    const int wave = tid >> 5;
    const int lhalf = lane >> 4;       // 0 or 1
    const int lrow  = lane & 15;

    const int n0 = blockIdx.x * 128;   // hw tile base
    const int o0 = blockIdx.y * 256;   // cout tile base
    const int bi = blockIdx.z;

    const int waveM = wave & 3;        // 4 waves over M -> 64 rows each
    const int waveN = wave >> 2;       // 2 waves over N -> 64 cols each
    const int wm0 = waveM * 64;
    const int wn0 = waveN * 64;

    // per-thread global-load coordinates
    const int arow = tid;                   // A: 256 rows, 1 thr/row, 32 halves
    const int bn   = tid & 127;             // B: column n within tile
    const int bkh  = tid >> 7;              // 0/1 -> k sub-half of 16
    const int pos  = n0 + bn;
    const int ph   = pos >> 6;
    const int pw   = pos & 63;

    v8f acc[4][4];
    #pragma unroll
    for (int m = 0; m < 4; ++m)
        #pragma unroll
        for (int q = 0; q < 4; ++q)
            acc[m][q] = (v8f)0.0f;

    const _Float16* xb_base = xh + (size_t)bi * CIN * NPOS;

    for (int r = 0; r < 9; ++r) {
        const int dh = r / 3 - 1;
        const int dw = r % 3 - 1;
        const int hs = ph + dh;
        const int ws2 = pw + dw;
        const bool inb = ((unsigned)hs < (unsigned)HW) && ((unsigned)ws2 < (unsigned)HW);
        const _Float16* Ag = wm + (((size_t)bi * KS + r) * COUT) * CIN;
        const _Float16* xsrc0 = xb_base + hs * HW + ws2;   // + i*NPOS per channel

        for (int ki = 0; ki < CIN / 32; ++ki) {
            // ---- stage tiles from global into registers ----
            const _Float16* asrc = Ag + (size_t)(o0 + arow) * CIN + ki * 32;
            v8h a0 = *(const v8h*)(asrc);
            v8h a1 = *(const v8h*)(asrc + 8);
            v8h a2 = *(const v8h*)(asrc + 16);
            v8h a3 = *(const v8h*)(asrc + 24);

            _Float16 tmp[16];
            const int ibase = ki * 32 + bkh * 16;
            if (inb) {
                const _Float16* src = xsrc0 + (size_t)ibase * NPOS;
                #pragma unroll
                for (int j = 0; j < 16; ++j) tmp[j] = src[(size_t)j * NPOS];
            } else {
                #pragma unroll
                for (int j = 0; j < 16; ++j) tmp[j] = (_Float16)0.0f;
            }

            __syncthreads();   // previous iteration's fragment reads done

            {
                v8h* adst = (v8h*)&As[arow][0];
                adst[0] = a0; adst[1] = a1; adst[2] = a2; adst[3] = a3;
            }
            {
                v8h blo, bhi;
                #pragma unroll
                for (int j = 0; j < 8; ++j) { blo[j] = tmp[j]; bhi[j] = tmp[j + 8]; }
                v8h* bdst = (v8h*)&Bs[bn][bkh * 16];
                bdst[0] = blo;
                bdst[1] = bhi;
            }

            __syncthreads();

            // ---- load fragments per CDNA5 per-lane layouts ----
            // A 16x32 f16: lane(l<16) row=l : K 0-7 (v0-3), 16-23 (v4-7)
            //              lane(l>=16) row=l-16 : K 8-15, 24-31
            v16h af[4];
            #pragma unroll
            for (int m = 0; m < 4; ++m) {
                const _Float16* p = &As[wm0 + m * 16 + lrow][lhalf * 8];
                v8h lo = *(const v8h*)p;
                v8h hi = *(const v8h*)(p + 16);
                af[m] = __builtin_shufflevector(lo, hi, 0,1,2,3,4,5,6,7,
                                                         8,9,10,11,12,13,14,15);
            }
            // B 32x16 f16: lane(l<16) col=l : K 0-7, 16-23; lane(l>=16) col=l-16 : K 8-15, 24-31
            v16h bf[4];
            #pragma unroll
            for (int q = 0; q < 4; ++q) {
                const _Float16* p = &Bs[wn0 + q * 16 + lrow][lhalf * 8];
                v8h lo = *(const v8h*)p;
                v8h hi = *(const v8h*)(p + 16);
                bf[q] = __builtin_shufflevector(lo, hi, 0,1,2,3,4,5,6,7,
                                                         8,9,10,11,12,13,14,15);
            }

            #pragma unroll
            for (int m = 0; m < 4; ++m)
                #pragma unroll
                for (int q = 0; q < 4; ++q)
                    acc[m][q] = __builtin_amdgcn_wmma_f32_16x16x32_f16(
                        false, af[m], false, bf[q], (short)0, acc[m][q],
                        false, false);
        }
    }

    // ---- store C: element (vgpr j, lane): M = j + 8*lhalf, N = lrow ----
    #pragma unroll
    for (int m = 0; m < 4; ++m) {
        #pragma unroll
        for (int q = 0; q < 4; ++q) {
            const int om = o0 + wm0 + m * 16 + lhalf * 8;
            const int nn = n0 + wn0 + q * 16 + lrow;
            #pragma unroll
            for (int j = 0; j < 8; ++j) {
                out[((size_t)bi * COUT + om + j) * NPOS + nn] =
                    acc[m][q][j] + bias[om + j];
            }
        }
    }
}

// ---------------------------------------------------------------------------
extern "C" void kernel_launch(void* const* d_in, const int* in_sizes, int n_in,
                              void* d_out, int out_size, void* d_ws, size_t ws_size,
                              hipStream_t stream) {
    (void)in_sizes; (void)n_in; (void)out_size; (void)ws_size;
    const float* x    = (const float*)d_in[0];   // [8,512,64,64]
    const float* s    = (const float*)d_in[1];   // [8,512]
    const float* w    = (const float*)d_in[2];   // [512,512,3,3]
    const float* bias = (const float*)d_in[3];   // [512]
    float* out = (float*)d_out;                  // [8,512,64,64]

    _Float16* xh = (_Float16*)d_ws;              // 33.5 MB
    _Float16* wm = xh + XH_HALFS;                // 37.7 MB

    // 1) modulated+demodulated f16 weights, tap-major
    modulate_kernel<<<dim3(B_ * COUT), dim3(256), 0, stream>>>(w, s, wm);
    // 2) x -> f16
    {
        const size_t nelem = (size_t)B_ * CIN * NPOS;   // 16,777,216
        const int blocks = (int)(nelem / (256 * 4));
        cvt_x_kernel<<<dim3(blocks), dim3(256), 0, stream>>>(x, xh);
    }
    // 3) implicit GEMM conv
    modconv_wmma_kernel<<<dim3(NPOS / 128, COUT / 256, B_), dim3(256), 0, stream>>>(
        xh, wm, bias, out);
}
